// BatchTopKSAE_30846455120259
// MI455X (gfx1250) — matrix operand order repaired and checked by reading, not compiled, not run on hardware
//
#include <hip/hip_runtime.h>
#include <hip/hip_bf16.h>
#include <stdint.h>

#define ACT_DIM 768
#define DICT    16384
#define BATCH   2048
#define KTOP    64
#define KTOT    (KTOP * BATCH)   // 131072

#define BM 64
#define BN 256
#define BK 32
#define KCHUNKS (ACT_DIM / BK)   // 24

typedef __attribute__((ext_vector_type(16))) __bf16 bf16x16;
typedef __attribute__((ext_vector_type(8)))  __bf16 bf16x8;
typedef __attribute__((ext_vector_type(8)))  float  f32x8;
typedef __attribute__((ext_vector_type(4)))  unsigned short us4;
typedef __attribute__((ext_vector_type(4)))  unsigned int   u32x4;
typedef __attribute__((ext_vector_type(8)))  int            i32x8;
typedef __attribute__((ext_vector_type(4)))  int            i32x4;

__device__ __forceinline__ unsigned short f32_to_bf16_rne(float f) {
  unsigned u = __float_as_uint(f);
  u += 0x7FFFu + ((u >> 16) & 1u);
  return (unsigned short)(u >> 16);
}

// ---------------------------------------------------------------------------
// Encoder GEMM: P = relu((x - b_dec) @ W_enc^T + b_enc), plus level-1 histogram
// Block tile 64x256, 8 waves, each wave computes 16x128 via 8 WMMA per K-chunk.
// ---------------------------------------------------------------------------
__global__ __launch_bounds__(256)
void sae_encode_wmma(const float* __restrict__ x,
                     const float* __restrict__ W_enc,
                     const float* __restrict__ b_enc,
                     const float* __restrict__ b_dec,
                     float* __restrict__ P,
                     unsigned* __restrict__ H1) {
  __shared__ __align__(64) unsigned short As[2][BM][BK];   // 4 KB x2
  __shared__ __align__(64) unsigned short Bs[2][BN][BK];   // 16 KB x2
  __shared__ unsigned hist[2048];                          // 8 KB

  const int tid = threadIdx.x;
  const int b0  = blockIdx.y * BM;   // batch-row tile
  const int f0  = blockIdx.x * BN;   // dict-col tile

  for (int i = tid; i < 2048; i += 256) hist[i] = 0u;

  // Issue ALL global loads first (max loads in flight), then convert + ds_store.
  auto load_chunk = [&](int kc, int bufi) {
    const int k0 = kc * BK;
    float4 av[2], dv[2], bv[8];
#pragma unroll
    for (int it = 0; it < 2; ++it) {
      int idx = tid + it * 256;
      int r = idx >> 3, c4 = (idx & 7) * 4;
      av[it] = *reinterpret_cast<const float4*>(&x[(size_t)(b0 + r) * ACT_DIM + k0 + c4]);
      dv[it] = *reinterpret_cast<const float4*>(&b_dec[k0 + c4]);
    }
#pragma unroll
    for (int it = 0; it < 8; ++it) {
      int idx = tid + it * 256;
      int r = idx >> 3, c4 = (idx & 7) * 4;
      bv[it] = *reinterpret_cast<const float4*>(&W_enc[(size_t)(f0 + r) * ACT_DIM + k0 + c4]);
    }
#pragma unroll
    for (int it = 0; it < 2; ++it) {
      int idx = tid + it * 256;
      int r = idx >> 3, c4 = (idx & 7) * 4;
      us4 p;
      p.x = f32_to_bf16_rne(av[it].x - dv[it].x);
      p.y = f32_to_bf16_rne(av[it].y - dv[it].y);
      p.z = f32_to_bf16_rne(av[it].z - dv[it].z);
      p.w = f32_to_bf16_rne(av[it].w - dv[it].w);
      *reinterpret_cast<us4*>(&As[bufi][r][c4]) = p;
    }
#pragma unroll
    for (int it = 0; it < 8; ++it) {
      int idx = tid + it * 256;
      int r = idx >> 3, c4 = (idx & 7) * 4;
      us4 p;
      p.x = f32_to_bf16_rne(bv[it].x);
      p.y = f32_to_bf16_rne(bv[it].y);
      p.z = f32_to_bf16_rne(bv[it].z);
      p.w = f32_to_bf16_rne(bv[it].w);
      *reinterpret_cast<us4*>(&Bs[bufi][r][c4]) = p;
    }
  };

  const int lane = tid & 31;
  const int wid  = tid >> 5;
  const int wm   = wid & 3;    // 4 M-subtiles of 16
  const int wn   = wid >> 2;   // 2 N-groups of 128
  const int l    = lane & 15;
  const int hi   = lane >> 4;  // 0 or 1

  f32x8 acc[8] = {};

  load_chunk(0, 0);
  __syncthreads();

  for (int kc = 0; kc < KCHUNKS; ++kc) {
    const int cur = kc & 1;
    if (kc + 1 < KCHUNKS) load_chunk(kc + 1, cur ^ 1);

    // A fragment (16x32 bf16): lane<16 holds M=l, K=0..15; lane>=16: K=16..31
    const unsigned short* arow = &As[cur][wm * 16 + l][0];
    bf16x8 alo = *reinterpret_cast<const bf16x8*>(arow + hi * 16);
    bf16x8 ahi = *reinterpret_cast<const bf16x8*>(arow + hi * 16 + 8);
    bf16x16 afrag = __builtin_shufflevector(alo, ahi,
        0, 1, 2, 3, 4, 5, 6, 7, 8, 9, 10, 11, 12, 13, 14, 15);

#pragma unroll
    for (int nt = 0; nt < 8; ++nt) {
      // B fragment (32x16): lane<16 holds N=l, K=0..15 (contiguous row slice of W_enc)
      const unsigned short* brow = &Bs[cur][wn * 128 + nt * 16 + l][hi * 16];
      bf16x16 bfrag = *reinterpret_cast<const bf16x16*>(brow);
      acc[nt] = __builtin_amdgcn_wmma_f32_16x16x32_bf16(
          false, afrag, false, bfrag, (short)0, acc[nt], false, false);
    }
    __syncthreads();
  }

  // Epilogue: bias + relu + store + histogram (C layout: N=lane&15, M=i+8*hi)
#pragma unroll
  for (int nt = 0; nt < 8; ++nt) {
    const int gn = f0 + wn * 128 + nt * 16 + l;
    const float be = b_enc[gn];
#pragma unroll
    for (int i = 0; i < 8; ++i) {
      const int gm = b0 + wm * 16 + hi * 8 + i;
      float v = acc[nt][i] + be;
      v = fmaxf(v, 0.0f);
      P[(size_t)gm * DICT + gn] = v;
      if (v > 0.0f) atomicAdd(&hist[__float_as_uint(v) >> 20], 1u);
    }
  }
  __syncthreads();
  for (int i = tid; i < 2048; i += 256) {
    const unsigned c = hist[i];
    if (c) atomicAdd(&H1[i], c);
  }
}

// ---------------------------------------------------------------------------
// Transpose W_dec [768 x 16384] -> WdT [16384 x 768] for coalesced decode
// ---------------------------------------------------------------------------
__global__ __launch_bounds__(256)
void transpose_wdec(const float* __restrict__ Wd, float* __restrict__ WdT) {
  __shared__ float tile[32][33];
  const int f0 = blockIdx.x * 32, d0 = blockIdx.y * 32;
  const int tx = threadIdx.x & 31, ty = threadIdx.x >> 5;
#pragma unroll
  for (int i = 0; i < 4; ++i)
    tile[ty + i * 8][tx] = Wd[(size_t)(d0 + ty + i * 8) * DICT + f0 + tx];
  __syncthreads();
#pragma unroll
  for (int i = 0; i < 4; ++i)
    WdT[(size_t)(f0 + ty + i * 8) * ACT_DIM + d0 + tx] = tile[tx][ty + i * 8];
}

// ---------------------------------------------------------------------------
// Level-1 threshold scan (single block): find bin T1 and residual count R1
// CTL[0]=T1 (0xFFFFFFFF => keep all positives), CTL[1]=R1, CTL[2]=thr bits
// ---------------------------------------------------------------------------
__global__ __launch_bounds__(256)
void scan_hist1(const unsigned* __restrict__ H, unsigned* __restrict__ CTL) {
  __shared__ unsigned h[2048];
  __shared__ unsigned cs[256];
  const int t = threadIdx.x;
  for (int i = t; i < 2048; i += 256) h[i] = H[i];
  __syncthreads();
  unsigned s = 0;
#pragma unroll
  for (int j = 0; j < 8; ++j) s += h[t * 8 + j];
  cs[t] = s;
  __syncthreads();
  if (t == 0) {
    const unsigned target = KTOT;
    unsigned cum = 0;
    int T1 = -1;
    for (int c = 255; c >= 0; --c) {
      if (cum + cs[c] >= target) {
        for (int j = 7; j >= 0; --j) {
          const unsigned v = h[c * 8 + j];
          if (cum + v >= target) { T1 = c * 8 + j; break; }
          cum += v;
        }
        break;
      }
      cum += cs[c];
    }
    if (T1 < 0) { CTL[0] = 0xFFFFFFFFu; CTL[1] = 0u; CTL[2] = 0u; }
    else        { CTL[0] = (unsigned)T1; CTL[1] = target - cum; CTL[2] = 0u; }
  }
}

// ---------------------------------------------------------------------------
// Level-2 histogram over bits[19:9] of values falling in bin T1
// ---------------------------------------------------------------------------
__global__ __launch_bounds__(256)
void hist_refine(const float* __restrict__ P, const unsigned* __restrict__ CTL,
                 unsigned* __restrict__ H2) {
  __shared__ unsigned h[2048];
  const int t = threadIdx.x;
  for (int i = t; i < 2048; i += 256) h[i] = 0u;
  __syncthreads();
  const unsigned T1 = CTL[0];
  if (T1 != 0xFFFFFFFFu) {
    const size_t N4 = (size_t)BATCH * DICT / 4;
    const float4* P4 = (const float4*)P;
    for (size_t i = (size_t)blockIdx.x * 256 + t; i < N4; i += (size_t)gridDim.x * 256) {
      const float4 v = P4[i];
      const float vs[4] = {v.x, v.y, v.z, v.w};
#pragma unroll
      for (int k = 0; k < 4; ++k) {
        const unsigned bits = __float_as_uint(vs[k]);
        if (vs[k] > 0.0f && (bits >> 20) == T1)
          atomicAdd(&h[(bits >> 9) & 2047u], 1u);
      }
    }
  }
  __syncthreads();
  for (int i = t; i < 2048; i += 256) {
    const unsigned c = h[i];
    if (c) atomicAdd(&H2[i], c);
  }
}

// ---------------------------------------------------------------------------
// Level-2 threshold scan: final threshold bits = (T1<<20)|(T2<<9)
// ---------------------------------------------------------------------------
__global__ __launch_bounds__(256)
void scan_hist2(const unsigned* __restrict__ H, unsigned* __restrict__ CTL) {
  __shared__ unsigned h[2048];
  __shared__ unsigned cs[256];
  const int t = threadIdx.x;
  for (int i = t; i < 2048; i += 256) h[i] = H[i];
  __syncthreads();
  unsigned s = 0;
#pragma unroll
  for (int j = 0; j < 8; ++j) s += h[t * 8 + j];
  cs[t] = s;
  __syncthreads();
  if (t == 0) {
    const unsigned T1 = CTL[0];
    if (T1 == 0xFFFFFFFFu) { CTL[2] = 0u; return; }
    const unsigned target = CTL[1];
    unsigned cum = 0;
    int T2 = -1;
    for (int c = 255; c >= 0; --c) {
      if (cum + cs[c] >= target) {
        for (int j = 7; j >= 0; --j) {
          const unsigned v = h[c * 8 + j];
          if (cum + v >= target) { T2 = c * 8 + j; break; }
          cum += v;
        }
        break;
      }
      cum += cs[c];
    }
    if (T2 < 0) T2 = 0;
    CTL[2] = (T1 << 20) | ((unsigned)T2 << 9);
  }
}

// ---------------------------------------------------------------------------
// Sparse decode: x_hat[b,:] = b_dec + sum_{kept f} P[b,f] * WdT[f,:]
// One block per batch row. The P row (64 KB, raw f32) is staged into LDS by
// the Tensor Data Mover: 2-D descriptor, data_size=4B, one 16384-wide row.
// TDM ignores EXEC, so only wave 0 issues it; tensorcnt + barrier publishes.
// ---------------------------------------------------------------------------
__global__ __launch_bounds__(256)
void sae_decode(const float* __restrict__ P, const float* __restrict__ WdT,
                const float* __restrict__ b_dec, const unsigned* __restrict__ CTL,
                float* __restrict__ out) {
  __shared__ __align__(128) float rowv[DICT];   // 64 KB of the 320 KB WGP LDS
  const int t = threadIdx.x;
  const int b = blockIdx.x;
  const unsigned thr = CTL[2];

#if __has_builtin(__builtin_amdgcn_tensor_load_to_lds)
  if (t < 32) {                                   // wave 0 only (uniform branch)
    const unsigned long long ga =
        (unsigned long long)(const void*)(P + (size_t)b * DICT);
    const unsigned lds = (unsigned)(unsigned long long)(const void*)&rowv[0];
    u32x4 g0;
    g0[0] = 1u;                                   // count=1 valid user D#
    g0[1] = lds;                                  // lds_addr (bytes)
    g0[2] = (unsigned)ga;                         // global_addr[31:0]
    g0[3] = (unsigned)(ga >> 32) | 0x80000000u;   // global_addr[56:32] | type=2
    i32x8 g1;
    g1[0] = 0x00020000;     // workgroup_mask=0, data_size=2 (4B), no flags
    g1[1] = 0x40000000;     // tensor_dim0 = 16384 (bits 79:48, low half here)
    g1[2] = 0x00010000;     // tensor_dim0 hi = 0, tensor_dim1 = 1
    g1[3] = 0x40000000;     // tensor_dim1 hi = 0, tile_dim0 = 16384
    g1[4] = 0x00000001;     // tile_dim1 = 1, tile_dim2 = 0
    g1[5] = 16384;          // tensor_dim0_stride (elements)
    g1[6] = 0;
    g1[7] = 0;
    i32x4 gz4 = {0, 0, 0, 0};
    i32x8 gz8 = {0, 0, 0, 0, 0, 0, 0, 0};
    __builtin_amdgcn_tensor_load_to_lds(g0, g1, gz4, gz4, gz8, 0);
#if __has_builtin(__builtin_amdgcn_s_wait_tensorcnt)
    __builtin_amdgcn_s_wait_tensorcnt(0);
#else
    asm volatile("s_wait_tensorcnt 0x0" ::: "memory");
#endif
  }
  __syncthreads();
#else
  for (int c = t; c < DICT; c += 256) rowv[c] = P[(size_t)b * DICT + c];
  __syncthreads();
#endif

  float a0 = b_dec[t], a1 = b_dec[t + 256], a2 = b_dec[t + 512];
  for (int j = 0; j < DICT; ++j) {
    const float v = rowv[j];                      // LDS broadcast, uniform branch
    if (v > 0.0f && __float_as_uint(v) >= thr) {
      const float* w = WdT + (size_t)j * ACT_DIM; // coalesced, L2-resident
      a0 = fmaf(v, w[t], a0);
      a1 = fmaf(v, w[t + 256], a1);
      a2 = fmaf(v, w[t + 512], a2);
    }
  }
  out[(size_t)b * ACT_DIM + t]       = a0;
  out[(size_t)b * ACT_DIM + t + 256] = a1;
  out[(size_t)b * ACT_DIM + t + 512] = a2;
}

// ---------------------------------------------------------------------------
extern "C" void kernel_launch(void* const* d_in, const int* in_sizes, int n_in,
                              void* d_out, int out_size, void* d_ws, size_t ws_size,
                              hipStream_t stream) {
  const float* x     = (const float*)d_in[0];
  const float* W_enc = (const float*)d_in[1];
  const float* b_enc = (const float*)d_in[2];
  const float* W_dec = (const float*)d_in[3];
  const float* b_dec = (const float*)d_in[4];
  float* out = (float*)d_out;

  const size_t P_BYTES = (size_t)BATCH * DICT * sizeof(float);   // 128 MiB
  float*    P   = (float*)d_ws;
  unsigned* H1  = (unsigned*)((char*)d_ws + P_BYTES);            // 2048 u32
  unsigned* H2  = H1 + 2048;                                     // 2048 u32
  unsigned* CTL = H2 + 2048;                                     // 3 u32 used
  float*    WdT = (float*)((char*)d_ws + P_BYTES + 16640);       // 48 MiB

  // Zero both histograms + control block every call (determinism).
  (void)hipMemsetAsync((void*)H1, 0, 16640, stream);

  transpose_wdec<<<dim3(DICT / 32, ACT_DIM / 32), 256, 0, stream>>>(W_dec, WdT);
  sae_encode_wmma<<<dim3(DICT / BN, BATCH / BM), 256, 0, stream>>>(
      x, W_enc, b_enc, b_dec, P, H1);
  scan_hist1<<<1, 256, 0, stream>>>(H1, CTL);
  hist_refine<<<2048, 256, 0, stream>>>(P, CTL, H2);
  scan_hist2<<<1, 256, 0, stream>>>(H2, CTL);
  sae_decode<<<BATCH, 256, 0, stream>>>(P, WdT, b_dec, CTL, out);
}
